// FoldingBlock_38972533244173
// MI455X (gfx1250) — compile-verified, matrix operand order checked by reading, not run on hardware
//
#include <hip/hip_runtime.h>

// ---------------------------------------------------------------------------
// MI455X (gfx1250) AlphaFold folding block.
// - All matmuls via v_wmma_f32_16x16x32_f16 (f16 inputs, f32 accumulate).
// - bgemm (triangle einsums / attention) uses the CDNA5 async path:
//   global_load_async_to_lds_b32 + s_wait_asynccnt with LDS double buffering,
//   overlapping HBM fetch with WMMA (workload is bandwidth-bound at 23.3TB/s).
// - gemm keeps fused epilogues (bias/sigmoid/gate/residual), synthesizes the
//   outer-product input on the fly (avoids a 201MB concat), and prefetches the
//   next weight tile (global_prefetch_b8).
// ---------------------------------------------------------------------------

typedef __attribute__((ext_vector_type(16))) _Float16 v16h;
typedef __attribute__((ext_vector_type(8)))  float    v8f;

#define DN   256
#define DNN  65536
#define DC   384
#define DP   128

#define GEMM_NONE 0
#define GEMM_SIG  1
#define GEMM_MUL  2
#define GEMM_ADD  3

struct GS { long long off, bs0, bs1, rs, cs; };

union FragU { v16h v; _Float16 h[16]; unsigned u[8]; };

// Fragment from f16 LDS row (stride 34 halfs). Lane layout per CDNA5 ISA
// 7.12.2 (16-bit A 16x32): VGPR t holds packed pair (k,k+1); fh selects halves.
__device__ __forceinline__ v16h load_frag(const _Float16* rowp, int fh) {
  FragU f;
#pragma unroll
  for (int t = 0; t < 4; ++t) {
    f.u[t]     = *(const unsigned*)(rowp + 2 * t + 8 * fh);
    f.u[t + 4] = *(const unsigned*)(rowp + 16 + 2 * t + 8 * fh);
  }
  return f.v;
}

// Fragment from f32 LDS row (stride 34 floats), converting to packed f16.
__device__ __forceinline__ v16h load_frag_f32(const float* rowp, int fh) {
  FragU f;
#pragma unroll
  for (int t = 0; t < 8; ++t) {
    int k = (t < 4) ? (2 * t + 8 * fh) : (16 + 2 * (t - 4) + 8 * fh);
    f.h[2 * t]     = (_Float16)rowp[k];
    f.h[2 * t + 1] = (_Float16)rowp[k + 1];
  }
  return f.v;
}

// CDNA5 async copy: one DWORD global -> LDS, tracked by ASYNCcnt.
__device__ __forceinline__ void async_copy_b32(void* lds_dst, const void* gsrc) {
  unsigned loff = (unsigned)(unsigned long long)lds_dst;   // LDS addr = addr[31:0]
  unsigned long long ga = (unsigned long long)gsrc;
  asm volatile("global_load_async_to_lds_b32 %0, %1, off"
               :: "v"(loff), "v"(ga) : "memory");
}
__device__ __forceinline__ void wait_async0() {
  asm volatile("s_wait_asynccnt 0x0" ::: "memory");
}

// ---------------------------------------------------------------------------
// General GEMM: O[M,Nc] = epilogue( A[M,K] @ W[K,Nc] + bias )
// aMode==1: A synthesized from s: row=(i*256+j), k<384 -> s[j,k]*s[i,k]
//           else s[j,k-384]-s[i,k-384]   (outer product / difference)
// premul: optional elementwise pre-multiplier on A (for (g*o) @ o_w).
// mode: NONE | SIG | MUL(*extra) | ADD(+extra), extra ld = Nc.
// ---------------------------------------------------------------------------
__global__ __launch_bounds__(128) void gemm_kernel(
    const float* __restrict__ A, const float* __restrict__ W,
    const float* __restrict__ bias, const float* __restrict__ extra,
    const float* __restrict__ premul, const float* __restrict__ sv,
    float* __restrict__ O, int M, int Nc, int K, int mode, int aMode) {
  __shared__ _Float16 As[64 * 34];
  __shared__ _Float16 Bs[64 * 34];
  const int tid  = threadIdx.x;
  const int m0   = blockIdx.y * 64, n0 = blockIdx.x * 64;
  const int wave = tid >> 5, lane = tid & 31;
  const int fm   = lane & 15, fh = lane >> 4;

  v8f acc[4] = {};

  for (int k0 = 0; k0 < K; k0 += 32) {
    __syncthreads();
    for (int idx = tid; idx < 64 * 32; idx += 128) {       // A tile (coalesced k)
      int r = idx >> 5, kk = idx & 31;
      int row = m0 + r, k = k0 + kk;
      float v = 0.f;
      if (row < M && k < K) {
        if (aMode == 0) {
          long long p = (long long)row * K + k;
          v = A[p];
          if (premul) v *= premul[p];
        } else {
          int i = row >> 8, j = row & 255;
          if (k < DC) v = sv[j * DC + k] * sv[i * DC + k];
          else { int c = k - DC; v = sv[j * DC + c] - sv[i * DC + c]; }
        }
      }
      As[r * 34 + kk] = (_Float16)v;
    }
    for (int idx = tid; idx < 64 * 32; idx += 128) {       // W tile (coalesced n)
      int kk = idx >> 6, n = idx & 63;
      int k = k0 + kk, col = n0 + n;
      float v = (k < K && col < Nc) ? W[(long long)k * Nc + col] : 0.f;
      Bs[n * 34 + kk] = (_Float16)v;
      if (k + 32 < K && col < Nc)                          // global_prefetch_b8
        __builtin_prefetch(&W[(long long)(k + 32) * Nc + col], 0, 1);
    }
    __syncthreads();

    v16h a = load_frag(&As[(wave * 16 + fm) * 34], fh);
#pragma unroll
    for (int c = 0; c < 4; ++c) {
      v16h b = load_frag(&Bs[(c * 16 + fm) * 34], fh);
      acc[c] = __builtin_amdgcn_wmma_f32_16x16x32_f16(
          false, a, false, b, (short)0, acc[c], false, false);
    }
  }

#pragma unroll
  for (int c = 0; c < 4; ++c) {
#pragma unroll
    for (int r = 0; r < 8; ++r) {
      int row = m0 + wave * 16 + fh * 8 + r;   // C/D: m = vgpr + 8*(lane>>4)
      int col = n0 + c * 16 + fm;
      if (row < M && col < Nc) {
        float y = acc[c][r];
        if (bias) y += bias[col];
        if (mode == GEMM_SIG)      y = 1.f / (1.f + __expf(-y));
        else if (mode == GEMM_MUL) y *= extra[(long long)row * Nc + col];
        else if (mode == GEMM_ADD) y += extra[(long long)row * Nc + col];
        O[(long long)row * Nc + col] = y;
      }
    }
  }
}

// ---------------------------------------------------------------------------
// Batched strided GEMM (async double-buffered):
//   batch b (b0=b/nbInner, b1=b%nbInner):
//   O[row,col] = scale * sum_k A(row,k)*B(col,k)  (+ Z(row,col) if hasZ)
// Tiles staged in LDS as f32 via global_load_async_to_lds_b32; next K-tile
// fetch overlaps WMMA on the current tile; f16 conversion at fragment build.
// ---------------------------------------------------------------------------
__global__ __launch_bounds__(128) void bgemm_kernel(
    const float* __restrict__ A, GS ga, const float* __restrict__ B, GS gb,
    const float* __restrict__ Z, GS gz, float* __restrict__ O, GS go,
    int M, int Nc, int K, int nbInner, float scale, int hasZ) {
  __shared__ float Asf[2][64 * 34];
  __shared__ float Bsf[2][64 * 34];
  const int tid  = threadIdx.x;
  const int m0   = blockIdx.y * 64, n0 = blockIdx.x * 64;
  const int bIdx = blockIdx.z;
  const int b0 = bIdx / nbInner, b1 = bIdx % nbInner;
  const long long aoff = ga.off + ga.bs0 * b0 + ga.bs1 * b1;
  const long long boff = gb.off + gb.bs0 * b0 + gb.bs1 * b1;
  const long long zoff = gz.off + gz.bs0 * b0 + gz.bs1 * b1;
  const long long ooff = go.off + go.bs0 * b0 + go.bs1 * b1;
  const int wave = tid >> 5, lane = tid & 31;
  const int fm   = lane & 15, fh = lane >> 4;

  auto fill = [&](int p, int k0) {
    for (int idx = tid; idx < 64 * 32; idx += 128) {
      int r = idx >> 5, kk = idx & 31;
      int row = m0 + r, k = k0 + kk;
      if (row < M && k < K)
        async_copy_b32(&Asf[p][r * 34 + kk], A + aoff + ga.rs * row + ga.cs * k);
      else
        Asf[p][r * 34 + kk] = 0.f;
      int col = n0 + r;
      if (col < Nc && k < K)
        async_copy_b32(&Bsf[p][r * 34 + kk], B + boff + gb.rs * col + gb.cs * k);
      else
        Bsf[p][r * 34 + kk] = 0.f;
    }
  };

  v8f acc[4] = {};

  fill(0, 0);
  wait_async0();
  __syncthreads();

  int p = 0;
  for (int k0 = 0; k0 < K; k0 += 32) {
    if (k0 + 32 < K) fill(p ^ 1, k0 + 32);   // prefetch next tile (async)

    v16h a = load_frag_f32(&Asf[p][(wave * 16 + fm) * 34], fh);
#pragma unroll
    for (int c = 0; c < 4; ++c) {
      v16h b = load_frag_f32(&Bsf[p][(c * 16 + fm) * 34], fh);
      acc[c] = __builtin_amdgcn_wmma_f32_16x16x32_f16(
          false, a, false, b, (short)0, acc[c], false, false);
    }

    wait_async0();       // next tile landed in LDS
    __syncthreads();     // all waves done reading buf p / zero-fills visible
    p ^= 1;
  }

#pragma unroll
  for (int c = 0; c < 4; ++c) {
#pragma unroll
    for (int r = 0; r < 8; ++r) {
      int row = m0 + wave * 16 + fh * 8 + r;
      int col = n0 + c * 16 + fm;
      if (row < M && col < Nc) {
        float y = scale * acc[c][r];
        if (hasZ) y += Z[zoff + gz.rs * row + gz.cs * col];
        O[ooff + go.rs * row + go.cs * col] = y;
      }
    }
  }
}

// LayerNorm over rows of 128 (one block per row, blockDim=128).
__global__ __launch_bounds__(128) void ln_kernel(
    const float* __restrict__ x, const float* __restrict__ g,
    const float* __restrict__ b, float* __restrict__ y) {
  __shared__ float red[128];
  long long row = blockIdx.x;
  int t = threadIdx.x;
  float v = x[row * DP + t];
  red[t] = v; __syncthreads();
  for (int s = 64; s > 0; s >>= 1) { if (t < s) red[t] += red[t + s]; __syncthreads(); }
  float mean = red[0] * (1.f / DP); __syncthreads();
  float d = v - mean;
  red[t] = d * d; __syncthreads();
  for (int s = 64; s > 0; s >>= 1) { if (t < s) red[t] += red[t + s]; __syncthreads(); }
  float var = red[0] * (1.f / DP);
  y[row * DP + t] = d * rsqrtf(var + 1e-5f) * g[t] + b[t];
}

// Softmax over contiguous rows of length 256 (one block per row).
__global__ __launch_bounds__(256) void softmax_kernel(float* __restrict__ x) {
  __shared__ float red[256];
  long long row = blockIdx.x;
  float* p = x + row * 256;
  int t = threadIdx.x;
  float v = p[t];
  red[t] = v; __syncthreads();
  for (int s = 128; s > 0; s >>= 1) { if (t < s) red[t] = fmaxf(red[t], red[t + s]); __syncthreads(); }
  float mx = red[0]; __syncthreads();
  float e = __expf(v - mx);
  red[t] = e; __syncthreads();
  for (int s = 128; s > 0; s >>= 1) { if (t < s) red[t] += red[t + s]; __syncthreads(); }
  p[t] = e / red[0];
}

// Tiled transpose: in[R,S] -> out[S,R]
__global__ void transpose_kernel(const float* __restrict__ in,
                                 float* __restrict__ out, int R, int S) {
  __shared__ float tile[32][33];
  int r0 = blockIdx.y * 32, s0 = blockIdx.x * 32;
  int tx = threadIdx.x, ty = threadIdx.y;
  for (int i = 0; i < 32; i += 8) {
    int r = r0 + ty + i, s = s0 + tx;
    if (r < R && s < S) tile[ty + i][tx] = in[(long long)r * S + s];
  }
  __syncthreads();
  for (int i = 0; i < 32; i += 8) {
    int s = s0 + ty + i, r = r0 + tx;
    if (r < R && s < S) out[(long long)s * R + r] = tile[tx][ty + i];
  }
}

// out[j,i,c] = in[i,j,c] for [256,256,128]
__global__ __launch_bounds__(256) void swap01_kernel(const float* __restrict__ in,
                                                     float* __restrict__ out) {
  long long idx = (long long)blockIdx.x * 256 + threadIdx.x;
  long long t = idx;
  int c = (int)(t & 127); t >>= 7;
  int j = (int)(t & 255); int i = (int)(t >> 8);
  out[(((long long)j << 8) + i) * DP + c] = in[idx];
}

__global__ __launch_bounds__(256) void add_kernel(float* __restrict__ dst,
                                                  const float* __restrict__ u) {
  long long idx = (long long)blockIdx.x * 256 + threadIdx.x;
  dst[idx] += u[idx];
}

// dst[i,j,c] += u[j,i,c]
__global__ __launch_bounds__(256) void add_t01_kernel(float* __restrict__ dst,
                                                      const float* __restrict__ u) {
  long long idx = (long long)blockIdx.x * 256 + threadIdx.x;
  long long t = idx;
  int c = (int)(t & 127); t >>= 7;
  int j = (int)(t & 255); int i = (int)(t >> 8);
  dst[idx] += u[(((long long)j << 8) + i) * DP + c];
}

// ---------------------------------------------------------------------------
extern "C" void kernel_launch(void* const* d_in, const int* in_sizes, int n_in,
                              void* d_out, int out_size, void* d_ws, size_t ws_size,
                              hipStream_t stream) {
  (void)in_sizes; (void)n_in; (void)out_size; (void)ws_size;
  auto in = [&](int i) { return (const float*)d_in[i]; };

  const float* single  = in(0);
  const float* pair_in = in(1);

  float* s_out   = (float*)d_out;                 // [256,384]
  float* pairbuf = (float*)d_out + 98304;         // [256,256,128] == x0 (B=1)

  const long long BIG = (long long)DNN * DP;      // 8388608
  float* ws = (float*)d_ws;
  float* B0 = ws + 0 * BIG; float* B1 = ws + 1 * BIG; float* B2 = ws + 2 * BIG;
  float* B3 = ws + 3 * BIG; float* B4 = ws + 4 * BIG; float* B5 = ws + 5 * BIG;
  float* B6 = ws + 6 * BIG; float* B7 = ws + 7 * BIG;
  float* sm = ws + 8 * BIG;
  float* bias12 = sm;            sm += (long long)DNN * 12;
  float* sq = sm;                sm += 98304;
  float* sk = sm;                sm += 98304;
  float* svv = sm;               sm += 98304;
  float* slog = sm;              sm += (long long)12 * DNN;
  float* so = sm;                sm += 98304;
  float* s1 = sm;                sm += 98304;
  float* bv = sm;                sm += (long long)DNN * 4;

  const GS GS0{0, 0, 0, 0, 0};
  const float iscale = 0.17677669529663687f;      // 1/sqrt(32)

  auto gemm = [&](const float* A, const float* W, const float* bias,
                  const float* extra, const float* premul, const float* sv,
                  float* O, int M, int Nc, int K, int mode, int aMode) {
    dim3 g((Nc + 63) / 64, (M + 63) / 64);
    gemm_kernel<<<g, 128, 0, stream>>>(A, W, bias, extra, premul, sv, O, M, Nc, K, mode, aMode);
  };
  auto bgemm = [&](const float* A, GS ga, const float* Bm, GS gb, const float* Z,
                   GS gz, float* O, GS go, int M, int Nc, int K, int nBatch,
                   int nbInner, float scale, int hasZ) {
    dim3 g((Nc + 63) / 64, (M + 63) / 64, nBatch);
    bgemm_kernel<<<g, 128, 0, stream>>>(A, ga, Bm, gb, Z, gz, O, go, M, Nc, K, nbInner, scale, hasZ);
  };
  auto transpose = [&](const float* inp, float* outp, int R, int S) {
    dim3 g((S + 31) / 32, (R + 31) / 32);
    transpose_kernel<<<g, dim3(32, 8), 0, stream>>>(inp, outp, R, S);
  };

  // ---- 0) pair output starts as pair_rep ----
  hipMemcpyAsync(pairbuf, pair_in, (size_t)BIG * sizeof(float),
                 hipMemcpyDeviceToDevice, stream);

  // ---- 1) bias = pair_rep @ to_b.w + to_b.b : [N,N,12] ----
  gemm(pair_in, in(2), in(3), nullptr, nullptr, nullptr, bias12, DNN, 12, DP, GEMM_NONE, 0);

  // ---- 2) single self-attention with pair bias ----
  gemm(single, in(4), in(5), nullptr, nullptr, nullptr, sq, DN, DC, DC, GEMM_NONE, 0);
  gemm(single, in(6), in(7), nullptr, nullptr, nullptr, sk, DN, DC, DC, GEMM_NONE, 0);
  gemm(single, in(8), in(9), nullptr, nullptr, nullptr, svv, DN, DC, DC, GEMM_NONE, 0);
  bgemm(sq, GS{0, 0, 32, DC, 1}, sk, GS{0, 0, 32, DC, 1},
        bias12, GS{0, 0, 1, (long long)DN * 12, 12},
        slog, GS{0, 0, (long long)DNN, DN, 1},
        DN, DN, 32, 12, 12, iscale, 1);
  softmax_kernel<<<12 * DN, 256, 0, stream>>>(slog);
  bgemm(slog, GS{0, 0, (long long)DNN, DN, 1}, svv, GS{0, 0, 32, 1, DC},
        nullptr, GS0, so, GS{0, 0, 32, DC, 1}, DN, 32, DN, 12, 12, 1.0f, 0);
  gemm(so, in(10), in(11), single, nullptr, nullptr, s1, DN, DC, DC, GEMM_ADD, 0);
  gemm(s1, in(12), in(13), s1, nullptr, nullptr, s_out, DN, DC, DC, GEMM_ADD, 0);

  // ---- 3) outer product/difference -> pair update (A synthesized from s) ----
  gemm(nullptr, in(14), in(15), pairbuf, nullptr, s_out, pairbuf, DNN, DP, 2 * DC, GEMM_ADD, 1);

  // ---- triangle multiplicative update ----
  auto tri_mult = [&](int p, bool outgoing) {
    ln_kernel<<<DNN, 128, 0, stream>>>(pairbuf, in(p + 0), in(p + 1), B0);
    gemm(B0, in(p + 2), in(p + 3), nullptr, nullptr, nullptr, B1, DNN, DP, DP, GEMM_SIG, 0);
    gemm(B0, in(p + 4), in(p + 5), B1, nullptr, nullptr, B2, DNN, DP, DP, GEMM_MUL, 0);   // a
    gemm(B0, in(p + 6), in(p + 7), nullptr, nullptr, nullptr, B1, DNN, DP, DP, GEMM_SIG, 0);
    gemm(B0, in(p + 8), in(p + 9), B1, nullptr, nullptr, B3, DNN, DP, DP, GEMM_MUL, 0);   // b
    gemm(B0, in(p + 10), in(p + 11), nullptr, nullptr, nullptr, B4, DNN, DP, DP, GEMM_SIG, 0); // g
    transpose(B2, B5, DNN, DP);  // at[c][i*N+k]
    transpose(B3, B6, DNN, DP);  // bt[c][j*N+k]
    GS ga = outgoing ? GS{0, 0, (long long)DNN, DN, 1}     // NT
                     : GS{0, 0, (long long)DNN, 1, DN};    // TN
    bgemm(B5, ga, B6, ga, nullptr, GS0, B2, GS{0, 0, (long long)DNN, DN, 1},
          DN, DN, DN, DP, DP, 1.0f, 0);
    transpose(B2, B3, DP, DNN);  // back to [N,N,C]
    ln_kernel<<<DNN, 128, 0, stream>>>(B3, in(p + 12), in(p + 13), B5);
    gemm(B5, in(p + 14), in(p + 15), B4, nullptr, nullptr, B6, DNN, DP, DP, GEMM_MUL, 0); // u
    add_kernel<<<(DNN * DP) / 256, 256, 0, stream>>>(pairbuf, B6);
  };
  tri_mult(16, true);   // tmo (outgoing)
  tri_mult(32, false);  // tmi (incoming)

  // ---- triangle attention ----
  auto tri_attn = [&](int p, const float* x, float* u_out) {
    ln_kernel<<<DNN, 128, 0, stream>>>(x, in(p + 0), in(p + 1), B0);
    gemm(B0, in(p + 2), nullptr, nullptr, nullptr, nullptr, B1, DNN, DP, DP, GEMM_NONE, 0); // q
    gemm(B0, in(p + 3), nullptr, nullptr, nullptr, nullptr, B2, DNN, DP, DP, GEMM_NONE, 0); // k
    gemm(B0, in(p + 4), nullptr, nullptr, nullptr, nullptr, B3, DNN, DP, DP, GEMM_NONE, 0); // v
    gemm(B0, in(p + 5), nullptr, nullptr, nullptr, nullptr, bv, DNN, 4, DP, GEMM_NONE, 0);  // bias
    gemm(B0, in(p + 6), in(p + 7), nullptr, nullptr, nullptr, B4, DNN, DP, DP, GEMM_SIG, 0); // g
    for (int i0 = 0; i0 < DN; i0 += 32) {   // 32 i-rows per chunk: 128 (i,h) batches
      long long qb = (long long)i0 * DN * DP;
      bgemm(B1, GS{qb, (long long)DN * DP, 32, DP, 1},
            B2, GS{qb, (long long)DN * DP, 32, DP, 1},
            bv, GS{0, 0, 1, (long long)DN * 4, 4},
            B5, GS{0, (long long)4 * DNN, DNN, DN, 1},
            DN, DN, 32, 128, 4, iscale, 1);
      softmax_kernel<<<128 * DN, 256, 0, stream>>>(B5);
      bgemm(B5, GS{0, (long long)4 * DNN, DNN, DN, 1},
            B3, GS{qb, (long long)DN * DP, 32, 1, DP},
            nullptr, GS0,
            B6, GS{qb, (long long)DN * DP, 32, DP, 1},
            DN, 32, DN, 128, 4, 1.0f, 0);
    }
    gemm(B6, in(p + 8), in(p + 9), nullptr, B4, nullptr, u_out, DNN, DP, DP, GEMM_NONE, 0);
  };

  tri_attn(48, pairbuf, B5);                                       // tas
  add_kernel<<<(DNN * DP) / 256, 256, 0, stream>>>(pairbuf, B5);

  swap01_kernel<<<(DNN * DP) / 256, 256, 0, stream>>>(pairbuf, B7); // x^T
  tri_attn(58, B7, B5);
  add_t01_kernel<<<(DNN * DP) / 256, 256, 0, stream>>>(pairbuf, B5);
}
